// GAT_44272522887474
// MI455X (gfx1250) — matrix operand order repaired and checked by reading, compiled, not run on hardware
//
#include <hip/hip_runtime.h>
#include <math.h>

typedef __attribute__((ext_vector_type(16))) _Float16 v16h;
typedef __attribute__((ext_vector_type(8)))  float    v8f;

#define TPB 256
#define NCLS 40

// One-shot weight prep: Bt[n][k] = f16(B[k][n]) for n < Ncol, else 0.
// Bt is [NcolPad x K], column-packed so a GEMM lane reads contiguous K.
__global__ void conv_bt(const float* __restrict__ B, _Float16* __restrict__ Bt,
                        int K, int Ncol, int NcolPad) {
  const int idx = blockIdx.x * blockDim.x + threadIdx.x;
  if (idx >= NcolPad * K) return;
  const int n = idx / K, k = idx - n * K;
  Bt[idx] = (n < Ncol) ? (_Float16)B[(size_t)k * Ncol + n] : (_Float16)0.0f;
}

// ---------------------------------------------------------------------------
// WMMA GEMM: C[M,NCOL] = A[M,K] * B[K,NCOL], f16 multiply / f32 accumulate via
// v_wmma_f32_16x16x32_f16 (wave32). One block = NT waves covers a 16-row
// stripe of C; wave w owns columns [16w,16w+16). A stripe staged in LDS (f16,
// padded stride -> conflict-free ds_load_b128 fragments). B comes from the
// pre-transposed f16 Bt: per k-step one contiguous 32B v16h load per lane.
//
// Fragment layouts per CDNA5 ISA 7.12.2 (wave32), half=lane>>4, q=lane&15:
//  A 16x32 f16 : row M=q;  VGPR j holds K pair k=16*half+2*j (this code's order, same as B)
//  B 32x16 f16 : col N=q;  VGPR j holds K pair k=16*half+2*j  (contiguous; this code's order, not the ISA table order)
//  C 16x16 f32 : VGPR r -> M=8*half+r, N=q
// ---------------------------------------------------------------------------
template <int K, int NT, int NCOL, bool NGUARD>
__global__ __launch_bounds__(32 * NT)
void gat_gemm_wmma(const float* __restrict__ A, const _Float16* __restrict__ Bt,
                   float* __restrict__ C, int M) {
  constexpr int LDK = K + 8;                 // pad: rows 4 banks apart
  __shared__ _Float16 sA[16 * LDK];

  const int mt  = blockIdx.x;
  const int tid = threadIdx.x;

  // cooperative load of the 16xK A stripe, fp32 -> f16, into padded LDS
  for (int i = tid; i < 16 * K; i += 32 * NT) {
    const int row = i / K, k = i - row * K;
    sA[row * LDK + k] = (_Float16)A[(size_t)(mt * 16 + row) * K + k];
  }
  __syncthreads();

  const int lane = tid & 31;
  const int wid  = tid >> 5;
  const int half = lane >> 4;
  const int q    = lane & 15;
  const int n    = wid * 16 + q;             // C column this lane owns
  const _Float16* pB = Bt + (size_t)n * K;   // column n of B, contiguous in K
  const _Float16* pA = sA + q * LDK;

  v8f acc = {};
#pragma unroll
  for (int k0 = 0; k0 < K; k0 += 32) {
    v16h a;
#pragma unroll
    for (int j = 0; j < 8; ++j) {            // two aligned 16B runs -> b128
      const int kA = k0 + half * 16 + j * 2;
      a[2 * j]     = pA[kA];
      a[2 * j + 1] = pA[kA + 1];
    }
    const v16h b = *(const v16h*)(pB + k0 + 16 * half);  // 32B aligned
    acc = __builtin_amdgcn_wmma_f32_16x16x32_f16(
        /*neg_a=*/false, a, /*neg_b=*/false, b,
        /*c_mod=*/(short)0, acc, /*reuse_a=*/false, /*reuse_b=*/false);
  }

  // M multiple of 16 -> no row guard; NCOL compile-time -> immediate offsets
  const bool nok = !NGUARD || (n < NCOL);
  float* pC = C + (size_t)(mt * 16 + half * 8) * NCOL + n;
#pragma unroll
  for (int r = 0; r < 8; ++r) {
    if (nok) pC[r * NCOL] = acc[r];
  }
}

// Per-(node,head) attention projections: a_src/a_dst = <h[n,h,:], att[h,:]>
__global__ void attn_node(const float* __restrict__ hm,
                          const float* __restrict__ att_s,
                          const float* __restrict__ att_d,
                          float* __restrict__ asrc, float* __restrict__ adst,
                          int nnodes, int heads, int ch) {
  const int idx = blockIdx.x * blockDim.x + threadIdx.x;
  if (idx >= nnodes * heads) return;
  const int n = idx / heads, h = idx - n * heads;
  const float* row = hm + (size_t)n * heads * ch + (size_t)h * ch;
  float s = 0.f, d = 0.f;
  for (int c = 0; c < ch; ++c) {
    const float v = row[c];
    s += v * att_s[h * ch + c];
    d += v * att_d[h * ch + c];
  }
  asrc[idx] = s;
  adst[idx] = d;
}

// monotone uint encoding of float for atomicMax-based segment max
__device__ __forceinline__ unsigned f2mono(float f) {
  unsigned u = __float_as_uint(f);
  return (u & 0x80000000u) ? ~u : (u | 0x80000000u);
}
__device__ __forceinline__ float mono2f(unsigned e) {
  return __uint_as_float((e & 0x80000000u) ? (e & 0x7FFFFFFFu) : ~e);
}

__device__ __forceinline__ void edge_sd(const long long* __restrict__ ei,
                                        int e, int E, int& s, int& d) {
  if (e < E) { s = (int)ei[e]; d = (int)ei[(size_t)E + e]; }
  else       { s = d = e - E; }                       // self-loop
}

// Pass A: logits = leaky_relu(a_src[s]+a_dst[d], 0.2); segment max over dst
__global__ void edge_logit_max(const long long* __restrict__ ei,
                               const float* __restrict__ asrc,
                               const float* __restrict__ adst,
                               float* __restrict__ logits,
                               unsigned* __restrict__ maxb,
                               int E, int etot, int heads) {
  const int idx = blockIdx.x * blockDim.x + threadIdx.x;
  if (idx >= etot * heads) return;
  const int e = idx / heads, h = idx - e * heads;
  int s, d; edge_sd(ei, e, E, s, d);
  float l = asrc[s * heads + h] + adst[d * heads + h];
  l = (l > 0.f) ? l : 0.2f * l;
  logits[idx] = l;
  atomicMax(&maxb[d * heads + h], f2mono(l));
}

// Pass B: w = exp(logit - max[dst]); denom[dst] += w (w overwrites logits buf)
__global__ void edge_exp(const long long* __restrict__ ei,
                         float* __restrict__ wbuf,
                         const unsigned* __restrict__ maxb,
                         float* __restrict__ denom,
                         int E, int etot, int heads) {
  const int idx = blockIdx.x * blockDim.x + threadIdx.x;
  if (idx >= etot * heads) return;
  const int e = idx / heads, h = idx - e * heads;
  int s, d; edge_sd(ei, e, E, s, d);
  const float w = expf(wbuf[idx] - mono2f(maxb[d * heads + h]));
  wbuf[idx] = w;
  atomicAdd(&denom[d * heads + h], w);
}

// Pass C: alpha = w/denom; out[dst,h,:] += h[src,h,:] * alpha
__global__ void edge_aggr(const long long* __restrict__ ei,
                          const float* __restrict__ wbuf,
                          const float* __restrict__ denom,
                          const float* __restrict__ hm,
                          float* __restrict__ outm,
                          int E, int etot, int heads, int ch) {
  const int idx = blockIdx.x * blockDim.x + threadIdx.x;
  if (idx >= etot * heads) return;
  const int e = idx / heads, h = idx - e * heads;
  int s, d; edge_sd(ei, e, E, s, d);
  const float alpha = wbuf[idx] / (denom[d * heads + h] + 1e-16f);
  const float* hrow = hm   + (size_t)s * heads * ch + (size_t)h * ch;
  float*       orow = outm + (size_t)d * heads * ch + (size_t)h * ch;
  for (int c = 0; c < ch; ++c) atomicAdd(&orow[c], hrow[c] * alpha);
}

// in-place bias add + optional ELU
__global__ void bias_act(float* __restrict__ a, const float* __restrict__ bias,
                         int total, int width, int do_elu) {
  const int idx = blockIdx.x * blockDim.x + threadIdx.x;
  if (idx >= total) return;
  float v = a[idx] + bias[idx % width];
  if (do_elu) v = (v > 0.f) ? v : (expf(v) - 1.f);
  a[idx] = v;
}

// final: out = log_softmax(aggr + b2) per node
__global__ void final_lsm(const float* __restrict__ aggr,
                          const float* __restrict__ b2,
                          float* __restrict__ out, int nnodes, int ncls) {
  const int i = blockIdx.x * blockDim.x + threadIdx.x;
  if (i >= nnodes) return;
  float v[NCLS];
  float mx = -3.4e38f;
  for (int c = 0; c < ncls; ++c) {
    v[c] = aggr[(size_t)i * ncls + c] + b2[c];
    mx = fmaxf(mx, v[c]);
  }
  float sum = 0.f;
  for (int c = 0; c < ncls; ++c) sum += expf(v[c] - mx);
  const float ls = logf(sum);
  for (int c = 0; c < ncls; ++c) out[(size_t)i * ncls + c] = v[c] - mx - ls;
}

extern "C" void kernel_launch(void* const* d_in, const int* in_sizes, int n_in,
                              void* d_out, int out_size, void* d_ws, size_t ws_size,
                              hipStream_t stream) {
  const float*     x   = (const float*)d_in[0];
  const long long* ei  = (const long long*)d_in[1];   // int64 [2,E] flat
  const float*     W1  = (const float*)d_in[2];
  const float*     as1 = (const float*)d_in[3];
  const float*     ad1 = (const float*)d_in[4];
  const float*     b1  = (const float*)d_in[5];
  const float*     W2  = (const float*)d_in[6];
  const float*     as2 = (const float*)d_in[7];
  const float*     ad2 = (const float*)d_in[8];
  const float*     b2  = (const float*)d_in[9];
  float* out = (float*)d_out;

  const int F = 256, HD = 8, CH = 16, D1 = 128;       // heads*hid = 128
  const int NCP2 = 48;                                // layer-2 cols padded
  const int N  = in_sizes[0] / F;                     // 50000 (divisible by 16)
  const int E  = in_sizes[1] / 2;                     // 800000
  const int ET = E + N;                               // + self-loops

  // carve workspace (all offsets 256B aligned)
  char* p = (char*)d_ws;
  auto carve = [&](size_t bytes) {
    char* r = p; p += (bytes + 255) & ~(size_t)255; return (void*)r;
  };
  float*    h1    = (float*)   carve((size_t)N * D1 * 4);
  float*    asrc1 = (float*)   carve((size_t)N * HD * 4);
  float*    adst1 = (float*)   carve((size_t)N * HD * 4);
  unsigned* maxb1 = (unsigned*)carve((size_t)N * HD * 4);
  float*    den1  = (float*)   carve((size_t)N * HD * 4);
  float*    wb1   = (float*)   carve((size_t)ET * HD * 4);
  float*    agg1  = (float*)   carve((size_t)N * D1 * 4);
  float*    h2    = (float*)   carve((size_t)N * NCLS * 4);
  float*    asrc2 = (float*)   carve((size_t)N * 4);
  float*    adst2 = (float*)   carve((size_t)N * 4);
  unsigned* maxb2 = (unsigned*)carve((size_t)N * 4);
  float*    den2  = (float*)   carve((size_t)N * 4);
  float*    wb2   = (float*)   carve((size_t)ET * 4);
  float*    agg2  = (float*)   carve((size_t)N * NCLS * 4);
  _Float16* Bt1   = (_Float16*)carve((size_t)D1 * F * 2);      // 128x256 f16
  _Float16* Bt2   = (_Float16*)carve((size_t)NCP2 * D1 * 2);   // 48x128 f16

  // per-call state init (mono-encoded 0 == minimum; harness won't re-poison)
  hipMemsetAsync(maxb1, 0, (size_t)N * HD * 4, stream);
  hipMemsetAsync(den1,  0, (size_t)N * HD * 4, stream);
  hipMemsetAsync(agg1,  0, (size_t)N * D1 * 4, stream);
  hipMemsetAsync(maxb2, 0, (size_t)N * 4, stream);
  hipMemsetAsync(den2,  0, (size_t)N * 4, stream);
  hipMemsetAsync(agg2,  0, (size_t)N * NCLS * 4, stream);

  // one-shot weight transpose/convert to column-packed f16
  conv_bt<<<(D1 * F + TPB - 1) / TPB, TPB, 0, stream>>>(W1, Bt1, F, D1, D1);
  conv_bt<<<(NCP2 * D1 + TPB - 1) / TPB, TPB, 0, stream>>>(W2, Bt2, D1, NCLS, NCP2);

  // ---------------- layer 1 (8 heads x 16) ----------------
  // block = 8 waves, one 16x128 C stripe per block; K=256 fully unrolled
  gat_gemm_wmma<256, 8, 128, false><<<N / 16, 256, 0, stream>>>(x, Bt1, h1, N);
  const int t1 = N * HD;
  attn_node<<<(t1 + TPB - 1) / TPB, TPB, 0, stream>>>(h1, as1, ad1, asrc1, adst1, N, HD, CH);
  const int te1 = ET * HD;
  edge_logit_max<<<(te1 + TPB - 1) / TPB, TPB, 0, stream>>>(ei, asrc1, adst1, wb1, maxb1, E, ET, HD);
  edge_exp      <<<(te1 + TPB - 1) / TPB, TPB, 0, stream>>>(ei, wb1, maxb1, den1, E, ET, HD);
  edge_aggr     <<<(te1 + TPB - 1) / TPB, TPB, 0, stream>>>(ei, wb1, den1, h1, agg1, E, ET, HD, CH);
  const int tn1 = N * D1;
  bias_act<<<(tn1 + TPB - 1) / TPB, TPB, 0, stream>>>(agg1, b1, tn1, D1, 1);

  // ---------------- layer 2 (1 head x 40) ----------------
  // block = 3 waves covering 48 padded cols (stores guarded to 40)
  gat_gemm_wmma<128, 3, NCLS, true><<<N / 16, 96, 0, stream>>>(agg1, Bt2, h2, N);
  attn_node<<<(N + TPB - 1) / TPB, TPB, 0, stream>>>(h2, as2, ad2, asrc2, adst2, N, 1, NCLS);
  edge_logit_max<<<(ET + TPB - 1) / TPB, TPB, 0, stream>>>(ei, asrc2, adst2, wb2, maxb2, E, ET, 1);
  edge_exp      <<<(ET + TPB - 1) / TPB, TPB, 0, stream>>>(ei, wb2, maxb2, den2, E, ET, 1);
  edge_aggr     <<<(ET + TPB - 1) / TPB, TPB, 0, stream>>>(ei, wb2, den2, h2, agg2, E, ET, 1, NCLS);
  final_lsm<<<(N + TPB - 1) / TPB, TPB, 0, stream>>>(agg2, b2, out, N, NCLS);
}
